// GNNEncoder_61314953117893
// MI455X (gfx1250) — matrix operand order repaired and checked by reading, not compiled, not run on hardware
//
#include <hip/hip_runtime.h>
#include <hip/hip_bf16.h>

typedef __attribute__((ext_vector_type(16))) _Float16 v16h;
typedef __attribute__((ext_vector_type(8)))  float    v8f;
typedef __attribute__((ext_vector_type(4)))  unsigned int v4u;

#define B_     8
#define N_     1024
#define DIN_   64
#define H_     128
#define L_     2
#define HEADS_ 4
#define HD_    32
#define ROWS_  (B_ * N_)

// ---------------------------------------------------------------------------
// WMMA helpers (gfx1250 wave32 layouts, CDNA5 ISA §7.12.2)
// ---------------------------------------------------------------------------

__device__ inline v8f wmma_f16(v16h a, v16h b, v8f c) {
  // D = A(16x32 f16) * B(32x16 f16) + C(16x16 f32)
  return __builtin_amdgcn_wmma_f32_16x16x32_f16(
      false, a, false, b, (short)0, c, false, false);
}

// A-matrix tile (16x32, f16). lane&15 = row M, halves j -> K = (j&7)+8*((j>>3)*2+g)
__device__ inline v16h load_A(const _Float16* base, int stride) {
  int lane = threadIdx.x & 31;
  int m = lane & 15, g = lane >> 4;
  const _Float16* p = base + (size_t)m * stride + g * 8;
  union { v16h v; v4u q[2]; } u;
  u.q[0] = *(const v4u*)(p);       // K = 8g .. 8g+7
  u.q[1] = *(const v4u*)(p + 16);  // K = 16+8g .. 16+8g+7
  return u.v;
}

// B-matrix tile (32x16, f16), loaded from B^T rows (i.e. rows of W / K / V^T).
// lane&15 = column N; halves j -> K = (j&7)+8*((j>>3)*2+g) (same as A).
__device__ inline v16h load_B(const _Float16* base, int stride) {
  int lane = threadIdx.x & 31;
  int n = lane & 15, g = lane >> 4;
  const _Float16* p = base + (size_t)n * stride + g * 8;
  union { v16h v; v4u q[2]; } u;
  u.q[0] = *(const v4u*)(p);
  u.q[1] = *(const v4u*)(p + 16);
  return u.v;
}

// B tile with only K=0..15 valid (K=16..31 zero-padded): halves 8..15 are zeros.
__device__ inline v16h load_B_k16(const _Float16* base, int stride) {
  int lane = threadIdx.x & 31;
  int n = lane & 15, g = lane >> 4;
  v4u z = {0u, 0u, 0u, 0u};
  union { v16h v; v4u q[2]; } u;
  u.q[0] = z; u.q[1] = z;
  {
    const _Float16* p = base + (size_t)n * stride + g * 8;
    u.q[0] = *(const v4u*)(p);
  }
  return u.v;
}

// A tile with only K=0..15 valid, sourced from a 16x16 row-major f16 buffer (LDS).
__device__ inline v16h load_A_k16(const _Float16* base) {
  int lane = threadIdx.x & 31;
  int m = lane & 15, g = lane >> 4;
  v4u z = {0u, 0u, 0u, 0u};
  union { v16h v; v4u q[2]; } u;
  u.q[0] = *(const v4u*)(base + m * 16 + g * 8);  // g0: K0..7, g1: K8..15
  u.q[1] = z;                                     // K>=16 padded with zero
  return u.v;
}

// ---------------------------------------------------------------------------
// Utility kernels
// ---------------------------------------------------------------------------

__global__ void cast_f32_f16_kernel(const float* __restrict__ src,
                                    _Float16* __restrict__ dst, int n) {
  int i = blockIdx.x * blockDim.x + threadIdx.x;
  if (i < n) dst[i] = (_Float16)src[i];
}

// v (ROWS x H, f32) -> vt[((b*HEADS+h)*HD + d) * N + n]  (per-head V^T, f16)
__global__ void transpose_v_kernel(const float* __restrict__ v,
                                   _Float16* __restrict__ vt) {
  int i = blockIdx.x * blockDim.x + threadIdx.x;
  if (i >= ROWS_ * H_) return;
  int row = i >> 7, c = i & 127;
  int b = row >> 10, nn = row & 1023;
  int h = c >> 5, d = c & 31;
  vt[((size_t)((b * HEADS_ + h) * HD_ + d)) * N_ + nn] = (_Float16)v[i];
}

// ---------------------------------------------------------------------------
// GEMM: C(MxN,f32) = A(MxK,f16) @ W^T (W is NxK,f16) + bias, optional ReLU.
// One wave per 32x32 output supertile: 2 A-tiles x 2 B-tiles -> 4 WMMAs per
// k-step from 4 tile loads (2x the flops/byte of a 16x16-per-wave design) and
// 4 independent accumulators to keep the matrix pipe busy.
// ---------------------------------------------------------------------------

__global__ void __launch_bounds__(128)
gemm_bias_kernel(const _Float16* __restrict__ A, const _Float16* __restrict__ W,
                 const float* __restrict__ bias, float* __restrict__ C,
                 int M, int N, int K, int relu) {
  int wave = blockIdx.x * (blockDim.x >> 5) + (threadIdx.x >> 5);
  int ntiles = N >> 5;
  if (wave >= (M >> 5) * ntiles) return;
  int tm = wave / ntiles, tn = wave % ntiles;

  const _Float16* a0 = A + (size_t)(tm * 32) * K;
  const _Float16* a1 = a0 + (size_t)16 * K;
  const _Float16* w0 = W + (size_t)(tn * 32) * K;
  const _Float16* w1 = w0 + (size_t)16 * K;

  v8f c00 = {}, c01 = {}, c10 = {}, c11 = {};
  for (int k0 = 0; k0 < K; k0 += 32) {
    v16h A0 = load_A(a0 + k0, K);
    v16h A1 = load_A(a1 + k0, K);
    v16h B0 = load_B(w0 + k0, K);
    v16h B1 = load_B(w1 + k0, K);
    c00 = wmma_f16(A0, B0, c00);
    c01 = wmma_f16(A0, B1, c01);
    c10 = wmma_f16(A1, B0, c10);
    c11 = wmma_f16(A1, B1, c11);
  }

  int lane = threadIdx.x & 31;
  int n = lane & 15, g = lane >> 4;
  float bv0 = bias[tn * 32 + n];
  float bv1 = bias[tn * 32 + 16 + n];

  auto store_tile = [&](v8f acc, int mbase, int nbase, float bv) {
#pragma unroll
    for (int r = 0; r < 8; ++r) {
      float v = acc[r] + bv;
      if (relu) v = v > 0.f ? v : 0.f;
      C[(size_t)(mbase + r + 8 * g) * N + nbase + n] = v;
    }
  };
  store_tile(c00, tm * 32,      tn * 32,      bv0);
  store_tile(c01, tm * 32,      tn * 32 + 16, bv1);
  store_tile(c10, tm * 32 + 16, tn * 32,      bv0);
  store_tile(c11, tm * 32 + 16, tn * 32 + 16, bv1);
}

// ---------------------------------------------------------------------------
// Fused flash-style masked multi-head attention.
// Block = 4 waves = 4 heads, handling one (batch, 16-row tile). adj tiles are
// staged in LDS once per block and shared by all heads; the next adj tile is
// prefetched (global_prefetch_b8) while the current one is consumed.
// ---------------------------------------------------------------------------

__global__ void __launch_bounds__(128)
attn_kernel(const _Float16* __restrict__ qh, const _Float16* __restrict__ kh,
            const _Float16* __restrict__ vt, const float* __restrict__ adj,
            _Float16* __restrict__ ah) {
  __shared__ float s_adj[256];
  __shared__ __align__(16) _Float16 s_p[4 * 256];

  int b    = blockIdx.x >> 6;          // 64 row tiles per graph
  int m0   = (blockIdx.x & 63) << 4;
  int wave = threadIdx.x >> 5;         // = head
  int lane = threadIdx.x & 31;
  int n = lane & 15, g = lane >> 4;
  const float scale = 0.17677669529663687f;  // 1/sqrt(32)

  // Q tile for this head: rows m0..m0+15, dims 0..31 (A layout, held in regs)
  v16h qa = load_A(qh + ((size_t)(b * N_ + m0) * H_ + wave * HD_), H_);

  v8f acc0 = {}, acc1 = {};
  float mrow[8], lrow[8];
#pragma unroll
  for (int r = 0; r < 8; ++r) { mrow[r] = -1e30f; lrow[r] = 0.f; }

  _Float16* myp = s_p + wave * 256;
  const float* adj_base = adj + (size_t)(b * N_ + m0) * N_;
  const int arow = threadIdx.x >> 4, acol = threadIdx.x & 15;

  for (int j0 = 0; j0 < N_; j0 += 16) {
    __syncthreads();  // previous iteration's s_adj consumers are done
    {
      int e = threadIdx.x;
      s_adj[e]       = adj_base[(size_t)arow * N_ + j0 + acol];
      s_adj[e + 128] = adj_base[(size_t)(arow + 8) * N_ + j0 + acol];
      if (j0 + 16 < N_) {  // prefetch next adj tile into the cache hierarchy
        __builtin_prefetch(adj_base + (size_t)arow * N_ + j0 + 16 + acol, 0, 0);
        __builtin_prefetch(adj_base + (size_t)(arow + 8) * N_ + j0 + 16 + acol, 0, 0);
      }
    }
    // K tile: B^T rows are rows of K (keys x dims) -> contiguous loads
    v16h kb = load_B(kh + ((size_t)(b * N_ + j0) * H_ + wave * HD_), H_);
    __syncthreads();  // s_adj ready

    v8f s = {};
    s = wmma_f16(qa, kb, s);  // S = Q K^T  (16x16, C layout)

#pragma unroll
    for (int r = 0; r < 8; ++r) {
      float a  = s_adj[(r + 8 * g) * 16 + n];
      float sv = (a != 0.f) ? s[r] * scale : -1e30f;
      // row max across 16 lanes of this half-wave
      float mx = sv;
      mx = fmaxf(mx, __shfl_xor(mx, 1, 32));
      mx = fmaxf(mx, __shfl_xor(mx, 2, 32));
      mx = fmaxf(mx, __shfl_xor(mx, 4, 32));
      mx = fmaxf(mx, __shfl_xor(mx, 8, 32));
      float mnew = fmaxf(mrow[r], mx);
      float f    = __expf(mrow[r] - mnew);
      float pv   = __expf(sv - mnew);
      float ps = pv;
      ps += __shfl_xor(ps, 1, 32);
      ps += __shfl_xor(ps, 2, 32);
      ps += __shfl_xor(ps, 4, 32);
      ps += __shfl_xor(ps, 8, 32);
      lrow[r] = lrow[r] * f + ps;
      mrow[r] = mnew;
      acc0[r] *= f;
      acc1[r] *= f;
      myp[(r + 8 * g) * 16 + n] = (_Float16)pv;  // P tile -> LDS (row-major)
    }
    asm volatile("s_wait_dscnt 0" ::: "memory");  // order LDS write->read in-wave

    v16h pa = load_A_k16(myp);  // C layout -> A layout via LDS, K padded to 32
    const _Float16* vbase =
        vt + ((size_t)(b * HEADS_ + wave) * HD_) * N_ + j0;
    v16h vb0 = load_B_k16(vbase, N_);             // dims 0..15
    v16h vb1 = load_B_k16(vbase + 16 * N_, N_);   // dims 16..31
    acc0 = wmma_f16(pa, vb0, acc0);
    acc1 = wmma_f16(pa, vb1, acc1);
  }

#pragma unroll
  for (int r = 0; r < 8; ++r) {
    float inv = 1.0f / fmaxf(lrow[r], 1e-20f);
    size_t base = (size_t)(b * N_ + m0 + r + 8 * g) * H_ + wave * HD_;
    ah[base + n]      = (_Float16)(acc0[r] * inv);
    ah[base + 16 + n] = (_Float16)(acc1[r] * inv);
  }
}

// ---------------------------------------------------------------------------
// h = relu(layernorm(g)) + h   (updates f32 h and its f16 mirror). Wave/row.
// ---------------------------------------------------------------------------

__global__ void __launch_bounds__(128)
ln_res_kernel(const float* __restrict__ gbuf, const float* __restrict__ gamma,
              const float* __restrict__ beta, float* __restrict__ hf,
              _Float16* __restrict__ hh) {
  int row  = blockIdx.x * (blockDim.x >> 5) + (threadIdx.x >> 5);
  int lane = threadIdx.x & 31;
  const float* gr = gbuf + (size_t)row * H_;
  float x0 = gr[lane], x1 = gr[lane + 32], x2 = gr[lane + 64], x3 = gr[lane + 96];
  float s  = x0 + x1 + x2 + x3;
  float s2 = x0 * x0 + x1 * x1 + x2 * x2 + x3 * x3;
#pragma unroll
  for (int m = 1; m < 32; m <<= 1) {
    s  += __shfl_xor(s, m, 32);
    s2 += __shfl_xor(s2, m, 32);
  }
  float mu   = s * (1.f / 128.f);
  float var  = s2 * (1.f / 128.f) - mu * mu;
  float rinv = rsqrtf(var + 1e-5f);
  float* hr = hf + (size_t)row * H_;
  _Float16* hhr = hh + (size_t)row * H_;
#pragma unroll
  for (int t = 0; t < 4; ++t) {
    int c = lane + t * 32;
    float y = (gr[c] - mu) * rinv * gamma[c] + beta[c];
    y = y > 0.f ? y : 0.f;
    float hn = y + hr[c];
    hr[c]  = hn;
    hhr[c] = (_Float16)hn;
  }
}

// ---------------------------------------------------------------------------
// Orchestration
// ---------------------------------------------------------------------------

extern "C" void kernel_launch(void* const* d_in, const int* in_sizes, int n_in,
                              void* d_out, int out_size, void* d_ws, size_t ws_size,
                              hipStream_t stream) {
  const float* x     = (const float*)d_in[0];
  const float* adj   = (const float*)d_in[1];
  const float* enc_w = (const float*)d_in[2];
  const float* enc_b = (const float*)d_in[3];
  const float* q_w   = (const float*)d_in[4];
  const float* q_b   = (const float*)d_in[5];
  const float* k_w   = (const float*)d_in[6];
  const float* k_b   = (const float*)d_in[7];
  const float* v_w   = (const float*)d_in[8];
  const float* v_b   = (const float*)d_in[9];
  const float* o_w   = (const float*)d_in[10];
  const float* o_b   = (const float*)d_in[11];
  const float* ln_g  = (const float*)d_in[12];
  const float* ln_b  = (const float*)d_in[13];
  const float* out_w = (const float*)d_in[14];
  const float* out_b = (const float*)d_in[15];

  char* ws = (char*)d_ws;
  size_t off = 0;
  auto alloc = [&](size_t bytes) -> void* {
    void* p = ws + off;
    off += (bytes + 255) & ~(size_t)255;
    return p;
  };

  _Float16* xh     = (_Float16*)alloc((size_t)ROWS_ * DIN_ * 2);
  _Float16* ew16   = (_Float16*)alloc((size_t)H_ * DIN_ * 2);
  _Float16* qw16   = (_Float16*)alloc((size_t)L_ * H_ * H_ * 2);
  _Float16* kw16   = (_Float16*)alloc((size_t)L_ * H_ * H_ * 2);
  _Float16* vw16   = (_Float16*)alloc((size_t)L_ * H_ * H_ * 2);
  _Float16* ow16   = (_Float16*)alloc((size_t)L_ * H_ * H_ * 2);
  _Float16* outw16 = (_Float16*)alloc((size_t)H_ * H_ * 2);
  float*    hf     = (float*)   alloc((size_t)ROWS_ * H_ * 4);
  _Float16* hh     = (_Float16*)alloc((size_t)ROWS_ * H_ * 2);
  float*    tmp    = (float*)   alloc((size_t)ROWS_ * H_ * 4);
  _Float16* qh     = (_Float16*)alloc((size_t)ROWS_ * H_ * 2);
  _Float16* kh     = (_Float16*)alloc((size_t)ROWS_ * H_ * 2);
  _Float16* vt     = (_Float16*)alloc((size_t)ROWS_ * H_ * 2);
  _Float16* ah     = (_Float16*)alloc((size_t)ROWS_ * H_ * 2);

  auto cast = [&](const float* s, _Float16* d, int n) {
    cast_f32_f16_kernel<<<(n + 255) / 256, 256, 0, stream>>>(s, d, n);
  };
  auto gemm = [&](const _Float16* A, const _Float16* W, const float* bias,
                  float* C, int M, int N, int K, int relu) {
    int waves = (M >> 5) * (N >> 5);
    gemm_bias_kernel<<<(waves + 3) / 4, 128, 0, stream>>>(A, W, bias, C, M, N, K, relu);
  };

  // f16 conversions of inputs/weights
  cast(x, xh, ROWS_ * DIN_);
  cast(enc_w, ew16, H_ * DIN_);
  cast(q_w, qw16, L_ * H_ * H_);
  cast(k_w, kw16, L_ * H_ * H_);
  cast(v_w, vw16, L_ * H_ * H_);
  cast(o_w, ow16, L_ * H_ * H_);
  cast(out_w, outw16, H_ * H_);

  // encoder: h = relu(x @ enc_w^T + enc_b)
  gemm(xh, ew16, enc_b, hf, ROWS_, H_, DIN_, 1);
  cast(hf, hh, ROWS_ * H_);

  for (int l = 0; l < L_; ++l) {
    const int wo = l * H_ * H_, bo = l * H_;
    gemm(hh, qw16 + wo, q_b + bo, tmp, ROWS_, H_, H_, 0);
    cast(tmp, qh, ROWS_ * H_);
    gemm(hh, kw16 + wo, k_b + bo, tmp, ROWS_, H_, H_, 0);
    cast(tmp, kh, ROWS_ * H_);
    gemm(hh, vw16 + wo, v_b + bo, tmp, ROWS_, H_, H_, 0);
    transpose_v_kernel<<<(ROWS_ * H_ + 255) / 256, 256, 0, stream>>>(tmp, vt);

    attn_kernel<<<B_ * (N_ / 16), 128, 0, stream>>>(qh, kh, vt, adj, ah);

    gemm(ah, ow16 + wo, o_b + bo, tmp, ROWS_, H_, H_, 0);
    ln_res_kernel<<<ROWS_ / 4, 128, 0, stream>>>(tmp, ln_g + bo, ln_b + bo, hf, hh);
  }

  // final: out = h @ out_w^T + out_b  (f32 output)
  gemm(hh, outw16, out_b, (float*)d_out, ROWS_, H_, H_, 0);
}
